// MambaPlus_72275709657328
// MI455X (gfx1250) — matrix operand order repaired and checked by reading, not compiled, not run on hardware
//
#include <hip/hip_runtime.h>
#include <hip/hip_bf16.h>

// ---------------- problem constants ----------------
#define D_MODEL 1024
#define D_STATE 16
#define D_CONV  4
#define D_INNER 2048            // EXPAND * D_MODEL
#define DT_RANK 64              // D_MODEL / 16
#define BATCH   2
#define SEQ     2048
#define MROWS   (BATCH * SEQ)   // 4096 token rows

// ---------------- CDNA5 vector types ----------------
typedef __attribute__((ext_vector_type(16))) __bf16 v16bf;
typedef __attribute__((ext_vector_type(8)))  float  v8f;
typedef __attribute__((ext_vector_type(4)))  unsigned int u32x4;
typedef __attribute__((ext_vector_type(8)))  int    i32x8;
typedef __attribute__((ext_vector_type(4)))  int    i32x4;

union FragBF {                  // 16 bf16 = 32 bytes
  v16bf v;
  uint4 q[2];
  unsigned short u[16];
};

__device__ __forceinline__ unsigned short f2bf(float f) {
  unsigned int u = __float_as_uint(f);
  u += 0x7FFFu + ((u >> 16) & 1u);       // round-to-nearest-even
  return (unsigned short)(u >> 16);
}
__device__ __forceinline__ float bf2f(unsigned short b) {
  return __uint_as_float(((unsigned int)b) << 16);
}

// ---------------- Tensor Data Mover descriptor builders (ISA 8.3/8.4) ----------------
#if defined(__has_builtin)
#  if __has_builtin(__builtin_amdgcn_tensor_load_to_lds)
#    define USE_TDM 1
#  endif
#endif
#ifndef USE_TDM
#  define USE_TDM 0
#endif

// D# group 0: count=1 | lds_addr | global_addr[56:0] | type=2
__device__ __forceinline__ u32x4 tdm_g0(unsigned lds_off, unsigned long long gaddr) {
  u32x4 g;
  g.x = 1u;                                            // count=1 (valid user D#)
  g.y = lds_off;                                       // lds_addr, bits 63:32
  g.z = (unsigned)(gaddr & 0xFFFFFFFFull);             // global_addr[31:0]
  g.w = (unsigned)((gaddr >> 32) & 0x1FFFFFFull) | (2u << 30);  // [56:32] | type=2
  return g;
}

// D# group 1: 2-D tile; data_size log2, tensor_dim0/1, tile_dim0/1, dim0 stride.
__device__ __forceinline__ i32x8 tdm_g1(unsigned dsz, unsigned long long td0,
                                        unsigned long long td1, unsigned tile0,
                                        unsigned tile1, unsigned long long s0) {
  unsigned long long q0 = ((unsigned long long)(dsz & 3u)) << 16;  // data_size @16
  q0 |= (td0 & 0xFFFFull) << 48;                                   // tensor_dim0[15:0] @48
  unsigned long long q1 = (td0 >> 16) & 0xFFFFull;                 // tensor_dim0[31:16] @64
  q1 |= (td1 & 0xFFFFFFFFull) << 16;                               // tensor_dim1 @80
  q1 |= ((unsigned long long)(tile0 & 0xFFFFu)) << 48;             // tile_dim0 @112
  unsigned long long q2 = (unsigned long long)(tile1 & 0xFFFFu);   // tile_dim1 @128 (tile_dim2=0)
  q2 |= (s0 & 0xFFFFFFFFull) << 32;                                // stride0[31:0] @160
  unsigned long long q3 = (s0 >> 32) & 0xFFFFull;                  // stride0[47:32] @192
  i32x8 g;
  g[0] = (int)(unsigned)q0; g[1] = (int)(unsigned)(q0 >> 32);
  g[2] = (int)(unsigned)q1; g[3] = (int)(unsigned)(q1 >> 32);
  g[4] = (int)(unsigned)q2; g[5] = (int)(unsigned)(q2 >> 32);
  g[6] = (int)(unsigned)q3; g[7] = (int)(unsigned)(q3 >> 32);
  return g;
}

// ---------------- bf16 WMMA GEMM: C[M,N] = A[M,K] @ BT[N,K]^T ----------------
// A row-major bf16 (lda=K), BT = B transposed, row-major [N][K] bf16 (ldb=K),
// C row-major f32 (ldc=N). Block tile 128x128, K-step 32, double-buffered LDS
// fed by the TDM (wave 0 issues tensor_load_to_lds for tile k+1 while all
// waves run WMMAs on tile k). Requires M % 128 == 0 and K % 32 == 0.
#define TM 128
#define TN 128
#define TK 32

__global__ __launch_bounds__(256)
void gemm_bf16_wmma(const unsigned short* __restrict__ A,
                    const unsigned short* __restrict__ BT,
                    float* __restrict__ C,
                    int M, int N, int K) {
  __shared__ unsigned short As[2][TM][TK];   // [m][k], 8 KB per buffer
  __shared__ unsigned short Bs[2][TN][TK];   // [n][k], 8 KB per buffer

  const int tid  = threadIdx.x;
  const int lane = tid & 31;
  const int wave = tid >> 5;
  const int wm   = wave >> 1;             // 0..3
  const int wn   = wave & 1;              // 0..1
  const int lm   = lane & 15;
  const int kh   = lane >> 4;

  const int rowBase = blockIdx.y * TM;
  const int colBase = blockIdx.x * TN;

  auto stage = [&](int buf, int k0) {
#if USE_TDM
    if (wave == 0) {
      i32x4 z4 = {0, 0, 0, 0};
      i32x8 z8 = {0, 0, 0, 0, 0, 0, 0, 0};
      // A tile: TM rows x TK contiguous K elements, row stride K
      u32x4 ga0 = tdm_g0((unsigned)(size_t)&As[buf][0][0],
                         (unsigned long long)(size_t)(A + (size_t)rowBase * K + k0));
      i32x8 ga1 = tdm_g1(1u, (unsigned long long)(K - k0),
                         (unsigned long long)(M - rowBase),
                         TK, TM, (unsigned long long)K);
      __builtin_amdgcn_tensor_load_to_lds(ga0, ga1, z4, z4, z8, 0);
      // B tile (from BT): TN rows x TK contiguous K elements, row stride K.
      // tensor_dim1 = N - colBase -> hardware zero-fill for the N=96 GEMM tail.
      u32x4 gb0 = tdm_g0((unsigned)(size_t)&Bs[buf][0][0],
                         (unsigned long long)(size_t)(BT + (size_t)colBase * K + k0));
      i32x8 gb1 = tdm_g1(1u, (unsigned long long)(K - k0),
                         (unsigned long long)(N - colBase),
                         TK, TN, (unsigned long long)K);
      __builtin_amdgcn_tensor_load_to_lds(gb0, gb1, z4, z4, z8, 0);
    }
#else
    // manual cooperative staging fallback
    for (int i = tid; i < TM * (TK / 8); i += 256) {
      int m = i >> 2, ks = (i & 3) * 8;
      const unsigned short* p = A + (size_t)(rowBase + m) * K + (k0 + ks);
      if (k0 + TK < K) __builtin_prefetch(p + TK, 0, 1);
      *(uint4*)&As[buf][m][ks] = *(const uint4*)p;
    }
    for (int i = tid; i < TN * (TK / 8); i += 256) {
      int n = i >> 2, ks = (i & 3) * 8;
      int gn = colBase + n;
      uint4 v = {0u, 0u, 0u, 0u};
      if (gn < N) {
        const unsigned short* p = BT + (size_t)gn * K + (k0 + ks);
        if (k0 + TK < K) __builtin_prefetch(p + TK, 0, 1);
        v = *(const uint4*)p;
      }
      *(uint4*)&Bs[buf][n][ks] = v;
    }
#endif
  };

  v8f acc[2][4] = {};
  int cur = 0;

  stage(0, 0);
  for (int k0 = 0; k0 < K; k0 += TK) {
#if USE_TDM
    if (wave == 0) __builtin_amdgcn_s_wait_tensorcnt((short)0);
#endif
    __syncthreads();                       // buf[cur] ready for all waves
    if (k0 + TK < K) stage(cur ^ 1, k0 + TK);   // overlap next-tile DMA with WMMAs

    // ---- fragment loads: two ds_load_b128 per 16x16 fragment ----
    FragBF a[2], b[4];
#pragma unroll
    for (int i = 0; i < 2; ++i) {
      int m = wm * 32 + i * 16 + lm;
      a[i].q[0] = *(const uint4*)&As[cur][m][kh * 8];       // K = kh*8 + 0..7
      a[i].q[1] = *(const uint4*)&As[cur][m][16 + kh * 8];  // K = 16 + kh*8 + 0..7
    }
#pragma unroll
    for (int j = 0; j < 4; ++j) {
      int n = wn * 64 + j * 16 + lm;
      b[j].q[0] = *(const uint4*)&Bs[cur][n][kh * 8];       // K = kh*8 + 0..7
      b[j].q[1] = *(const uint4*)&Bs[cur][n][16 + kh * 8];  // K = 16 + kh*8 + 0..7
    }

#pragma unroll
    for (int i = 0; i < 2; ++i)
#pragma unroll
      for (int j = 0; j < 4; ++j)
        acc[i][j] = __builtin_amdgcn_wmma_f32_16x16x32_bf16(
            false, a[i].v, false, b[j].v, (short)0, acc[i][j], false, false);

    cur ^= 1;
  }

  // ---- epilogue: VGPR g, lane L -> M = g + (L>=16)*8, N = L%16 ----
#pragma unroll
  for (int i = 0; i < 2; ++i) {
#pragma unroll
    for (int j = 0; j < 4; ++j) {
      int gm0 = rowBase + wm * 32 + i * 16 + kh * 8;
      int gn  = colBase + wn * 64 + j * 16 + lm;
      if (gn < N) {
#pragma unroll
        for (int g = 0; g < 8; ++g) {
          int gm = gm0 + g;
          if (gm < M) C[(size_t)gm * N + gn] = acc[i][j][g];
        }
      }
    }
  }
}

// ---------------- fp32 -> bf16 (strided source, layout-preserving) ----------------
__global__ __launch_bounds__(256)
void cvt_bf16_k(const float* __restrict__ src, unsigned short* __restrict__ dst,
                int rows, int cols, int ld) {
  long long i = (long long)blockIdx.x * 256 + threadIdx.x;
  if (i >= (long long)rows * cols) return;
  int r = (int)(i / cols), c = (int)(i % cols);
  dst[i] = f2bf(src[(size_t)r * ld + c]);
}

// ---------------- fp32 [K][N] -> bf16 transposed [N][K] (weights) ----------------
__global__ __launch_bounds__(256)
void cvt_bf16_t_k(const float* __restrict__ src, unsigned short* __restrict__ dst,
                  int K, int N) {
  long long i = (long long)blockIdx.x * 256 + threadIdx.x;
  if (i >= (long long)K * N) return;
  int n = (int)(i / K), k = (int)(i % K);
  dst[i] = f2bf(src[(size_t)k * N + n]);
}

// ---------------- causal depthwise conv (k=4) + bias + SiLU -> bf16 ----------------
__global__ __launch_bounds__(256)
void conv_silu_k(const float* __restrict__ xz,
                 const float* __restrict__ w,     // [D_INNER,1,4]
                 const float* __restrict__ bias,  // [D_INNER]
                 unsigned short* __restrict__ out) {
  int i = blockIdx.x * 256 + threadIdx.x;
  if (i >= MROWS * D_INNER) return;
  int d = i % D_INNER;
  int r = i / D_INNER;           // r = b*SEQ + s
  int s = r % SEQ;
  float acc = bias[d];
  const float* wd = w + d * D_CONV;
#pragma unroll
  for (int t = 0; t < D_CONV; ++t) {
    int ss = s - (D_CONV - 1) + t;
    if (ss >= 0)
      acc += wd[t] * xz[(size_t)(r - (D_CONV - 1) + t) * (2 * D_INNER) + d];
  }
  if (s + 1 < SEQ) __builtin_prefetch(&xz[(size_t)(r + 1) * (2 * D_INNER) + d], 0, 1);
  float sv = acc / (1.f + __expf(-acc));   // SiLU
  out[i] = f2bf(sv);
}

// ---------------- selective scan ----------------
// One lane per (batch, channel, state): 65536 lanes = 2048 wave32s.
// 16-lane shfl-xor butterfly for y = sum_n h*C; fused softplus, +u*D, *silu(z).
__global__ __launch_bounds__(256)
void scan_k(const float* __restrict__ dtp,        // [MROWS, D_INNER] pre-softplus
            const float* __restrict__ dt_b,       // [D_INNER]
            const float* __restrict__ A_log,      // [D_INNER, 16]
            const float* __restrict__ Dv,         // [D_INNER]
            const unsigned short* __restrict__ u_bf,  // [MROWS, D_INNER]
            const float* __restrict__ xdbl,       // [MROWS, 96]
            const float* __restrict__ xz,         // [MROWS, 4096]
            unsigned short* __restrict__ y_bf) {  // [MROWS, D_INNER]
  int g = blockIdx.x * 256 + threadIdx.x;
  int chan = g >> 4;
  int n = g & 15;
  if (chan >= BATCH * D_INNER) return;
  int b = chan / D_INNER;
  int d = chan % D_INNER;

  float A    = -__expf(A_log[d * D_STATE + n]);
  float Dd   = Dv[d];
  float bias = dt_b[d];
  float h = 0.f;

  for (int t = 0; t < SEQ; ++t) {
    int r = b * SEQ + t;
    float dtv = dtp[(size_t)r * D_INNER + d] + bias;
    dtv = (dtv > 20.f) ? dtv : log1pf(__expf(dtv));     // softplus
    float uu = bf2f(u_bf[(size_t)r * D_INNER + d]);
    float Bt = xdbl[r * 96 + DT_RANK + n];
    float Ct = xdbl[r * 96 + DT_RANK + D_STATE + n];
    h = __expf(dtv * A) * h + (dtv * uu) * Bt;
    float y = h * Ct;
    y += __shfl_xor(y, 8, 16);
    y += __shfl_xor(y, 4, 16);
    y += __shfl_xor(y, 2, 16);
    y += __shfl_xor(y, 1, 16);
    if (n == 0) {
      float z  = xz[(size_t)r * (2 * D_INNER) + D_INNER + d];
      float yv = (y + uu * Dd) * (z / (1.f + __expf(-z)));
      y_bf[(size_t)r * D_INNER + d] = f2bf(yv);
    }
  }
}

// ---------------- final gate: out = sig(fgl+b)*x + (1-sig)*nf ----------------
__global__ __launch_bounds__(256)
void combine_k(const float* __restrict__ x, const float* __restrict__ fgl,
               const float* __restrict__ fg_b, const float* __restrict__ nf,
               float* __restrict__ out) {
  int i = blockIdx.x * 256 + threadIdx.x;
  if (i >= MROWS * D_MODEL) return;
  float s = 1.f / (1.f + __expf(-(fgl[i] + fg_b[i % D_MODEL])));
  out[i] = s * x[i] + (1.f - s) * nf[i];
}

// ---------------- host-side orchestration ----------------
static inline unsigned blk(long long n) { return (unsigned)((n + 255) / 256); }

extern "C" void kernel_launch(void* const* d_in, const int* in_sizes, int n_in,
                              void* d_out, int out_size, void* d_ws, size_t ws_size,
                              hipStream_t stream) {
  (void)in_sizes; (void)n_in; (void)out_size; (void)ws_size;

  const float* x        = (const float*)d_in[0];   // [B,S,1024]
  const float* in_proj  = (const float*)d_in[1];   // [1024,4096]
  const float* conv_w   = (const float*)d_in[2];   // [2048,1,4]
  const float* conv_b   = (const float*)d_in[3];   // [2048]
  const float* x_proj   = (const float*)d_in[4];   // [2048,96]
  const float* dt_proj  = (const float*)d_in[5];   // [64,2048]
  const float* dt_b     = (const float*)d_in[6];   // [2048]
  const float* A_log    = (const float*)d_in[7];   // [2048,16]
  const float* Dv       = (const float*)d_in[8];   // [2048]
  const float* out_proj = (const float*)d_in[9];   // [2048,1024]
  const float* fg_w     = (const float*)d_in[10];  // [1024,1024]
  const float* fg_b     = (const float*)d_in[11];  // [1024]
  float* out = (float*)d_out;

  // workspace carve (~160 MB): bump allocator, 256B aligned
  char* p = (char*)d_ws;
  auto alloc = [&](size_t bytes) -> char* {
    char* r = p; p += (bytes + 255) & ~(size_t)255; return r;
  };
  unsigned short* x_bf     = (unsigned short*)alloc((size_t)MROWS * D_MODEL * 2);
  unsigned short* inproj_t = (unsigned short*)alloc((size_t)D_MODEL * 2 * D_INNER * 2);  // [4096][1024]
  unsigned short* xproj_t  = (unsigned short*)alloc((size_t)D_INNER * 96 * 2);           // [96][2048]
  unsigned short* dtproj_t = (unsigned short*)alloc((size_t)DT_RANK * D_INNER * 2);      // [2048][64]
  unsigned short* outprj_t = (unsigned short*)alloc((size_t)D_INNER * D_MODEL * 2);      // [1024][2048]
  unsigned short* fgw_t    = (unsigned short*)alloc((size_t)D_MODEL * D_MODEL * 2);      // [1024][1024]
  float*          xz       = (float*)alloc((size_t)MROWS * 2 * D_INNER * 4);
  unsigned short* xiconv_b = (unsigned short*)alloc((size_t)MROWS * D_INNER * 2);
  float*          xdbl     = (float*)alloc((size_t)MROWS * 96 * 4);
  unsigned short* dtr_b    = (unsigned short*)alloc((size_t)MROWS * DT_RANK * 2);
  float*          dtp      = (float*)alloc((size_t)MROWS * D_INNER * 4);
  unsigned short* y_bf     = (unsigned short*)alloc((size_t)MROWS * D_INNER * 2);
  // nf / fgl reuse the dtp region (dtp is dead after scan_k; same-stream order)
  float* nf  = dtp;
  float* fgl = dtp + (size_t)MROWS * D_MODEL;

  // 1) fp32 -> bf16 conversions: activations row-major, weights transposed
  cvt_bf16_k<<<blk((long long)MROWS * D_MODEL), 256, 0, stream>>>(x, x_bf, MROWS, D_MODEL, D_MODEL);
  cvt_bf16_t_k<<<blk((long long)D_MODEL * 2 * D_INNER), 256, 0, stream>>>(in_proj, inproj_t, D_MODEL, 2 * D_INNER);
  cvt_bf16_t_k<<<blk((long long)D_INNER * 96), 256, 0, stream>>>(x_proj, xproj_t, D_INNER, 96);
  cvt_bf16_t_k<<<blk((long long)DT_RANK * D_INNER), 256, 0, stream>>>(dt_proj, dtproj_t, DT_RANK, D_INNER);
  cvt_bf16_t_k<<<blk((long long)D_INNER * D_MODEL), 256, 0, stream>>>(out_proj, outprj_t, D_INNER, D_MODEL);
  cvt_bf16_t_k<<<blk((long long)D_MODEL * D_MODEL), 256, 0, stream>>>(fg_w, fgw_t, D_MODEL, D_MODEL);

  // 2) xz = x @ in_proj  [4096 x 4096, K=1024]
  gemm_bf16_wmma<<<dim3((2 * D_INNER) / TN, MROWS / TM), 256, 0, stream>>>(
      x_bf, inproj_t, xz, MROWS, 2 * D_INNER, D_MODEL);

  // 3) depthwise causal conv + SiLU -> bf16 u
  conv_silu_k<<<blk((long long)MROWS * D_INNER), 256, 0, stream>>>(xz, conv_w, conv_b, xiconv_b);

  // 4) x_dbl = u @ x_proj  [4096 x 96, K=2048]
  gemm_bf16_wmma<<<dim3(1, MROWS / TM), 256, 0, stream>>>(
      xiconv_b, xproj_t, xdbl, MROWS, 96, D_INNER);

  // 5) dt slice -> bf16, then dtp = dt @ dt_proj  [4096 x 2048, K=64]
  cvt_bf16_k<<<blk((long long)MROWS * DT_RANK), 256, 0, stream>>>(xdbl, dtr_b, MROWS, DT_RANK, 96);
  gemm_bf16_wmma<<<dim3(D_INNER / TN, MROWS / TM), 256, 0, stream>>>(
      dtr_b, dtproj_t, dtp, MROWS, D_INNER, DT_RANK);

  // 6) selective scan (fused softplus, +u*D, *silu(z)) -> y bf16
  scan_k<<<blk((long long)BATCH * D_INNER * D_STATE), 256, 0, stream>>>(
      dtp, dt_b, A_log, Dv, xiconv_b, xdbl, xz, y_bf);

  // 7) new_features = y @ out_proj ; fg_lin = x @ fg_w
  gemm_bf16_wmma<<<dim3(D_MODEL / TN, MROWS / TM), 256, 0, stream>>>(
      y_bf, outprj_t, nf, MROWS, D_MODEL, D_INNER);
  gemm_bf16_wmma<<<dim3(D_MODEL / TN, MROWS / TM), 256, 0, stream>>>(
      x_bf, fgw_t, fgl, MROWS, D_MODEL, D_MODEL);

  // 8) out = sigmoid(fg)*x + (1-sigmoid(fg))*new_features
  combine_k<<<blk((long long)MROWS * D_MODEL), 256, 0, stream>>>(x, fgl, fg_b, nf, out);
}
